// ExtractModel_42391327212111
// MI455X (gfx1250) — compile-verified, matrix operand order checked by reading, not run on hardware
//
#include <hip/hip_runtime.h>
#include <stdint.h>

#define B_     16
#define L_     64
#define F_     60
#define D_     256
#define U_     600
#define UPAD_  608
#define V_     10000
#define JMAX_  10
#define MINLEN 4
#define EDIM   7            // MAX_LEN - MIN_LEN + 1
#define NROWS  (B_ * L_)    // 1024
#define NEGV   (-1e9f)
#define THRESH 0.05f

typedef __attribute__((ext_vector_type(16))) __bf16 v16bf;
typedef __attribute__((ext_vector_type(8)))  float  v8f;

// ---- bf16 <-> f32 helpers (raw ushort storage, round-to-nearest-even) ----
__device__ __forceinline__ unsigned short f2bf(float f) {
    unsigned u = __float_as_uint(f);
    unsigned r = u + 0x7FFFu + ((u >> 16) & 1u);
    return (unsigned short)(r >> 16);
}
__device__ __forceinline__ float bf2f(unsigned short h) {
    return __uint_as_float(((unsigned)h) << 16);
}

// ---- order-preserving float <-> uint key (max over key == max over float) ----
__device__ __forceinline__ unsigned encf(float f) {
    unsigned u = __float_as_uint(f);
    return (u & 0x80000000u) ? ~u : (u | 0x80000000u);
}
__device__ __forceinline__ float decf(unsigned k) {
    unsigned u = (k & 0x80000000u) ? (k & 0x7FFFFFFFu) : ~k;
    return __uint_as_float(u);
}

// =====================================================================
// Kernel 1: row = (feats @ W), L2-normalize, store as bf16.
// Blocks 0..1023 -> word rows (x), blocks 1024..1623 -> unit rows.
// =====================================================================
__global__ __launch_bounds__(256) void embed_norm_kernel(
    const float* __restrict__ x,
    const float* __restrict__ unit_feats,
    const float* __restrict__ W,
    unsigned short* __restrict__ wn,
    unsigned short* __restrict__ un)
{
    __shared__ float s_feat[F_];
    __shared__ float s_red[256];
    const int row = blockIdx.x;
    const int tid = threadIdx.x;

    const float* src;
    unsigned short* dst;
    if (row < NROWS) { src = x + (size_t)row * F_;            dst = wn + (size_t)row * D_; }
    else             { src = unit_feats + (size_t)(row - NROWS) * F_;
                       dst = un + (size_t)(row - NROWS) * D_; }

    if (tid < F_) s_feat[tid] = src[tid];
    __syncthreads();

    float acc = 0.0f;
    #pragma unroll 10
    for (int k = 0; k < F_; ++k)
        acc = fmaf(s_feat[k], W[k * D_ + tid], acc);   // coalesced W column reads

    s_red[tid] = acc * acc;
    __syncthreads();
    for (int s = 128; s > 0; s >>= 1) {
        if (tid < s) s_red[tid] += s_red[tid + s];
        __syncthreads();
    }
    const float scale = 1.0f / (sqrtf(s_red[0]) + 1e-8f);
    dst[tid] = f2bf(acc * scale);
}

// =====================================================================
// Kernel 2: sim = wn(1024x256) @ un^T(256x600) via v_wmma_f32_16x16x32_bf16.
// One wave per 16x16 tile. Grid = (64 Mtiles, 38 Ntiles). N padded to 608
// in the workspace so padded B loads stay in-bounds; stores guarded @600.
// A (16x32 bf16) per-lane layout: lanes 0-15 row M=lane, K-chunks {0..7},
// {16..23}; lanes 16-31 row M=lane-16, K-chunks {8..15},{24..31}. B mirrors
// with N in place of M (B = un^T, so B column n == un row n).
// =====================================================================
__global__ __launch_bounds__(32) void sim_wmma_kernel(
    const unsigned short* __restrict__ wn,
    const unsigned short* __restrict__ un,
    float* __restrict__ sim)
{
    const int mTile = blockIdx.x;            // 0..63
    const int nTile = blockIdx.y;            // 0..37
    const int lane  = threadIdx.x;           // 0..31
    const int half  = (lane < 16) ? 0 : 8;   // K sub-chunk base within 32
    const int mrow  = mTile * 16 + (lane & 15);
    const int ncol  = nTile * 16 + (lane & 15);   // max 607 < UPAD_

    const uint32_t* arow = (const uint32_t*)(wn + (size_t)mrow * D_); // dword view
    const uint32_t* brow = (const uint32_t*)(un + (size_t)ncol * D_);

    v8f c = {};
    #pragma unroll
    for (int kk = 0; kk < D_; kk += 32) {
        union { uint4 q[2]; v16bf v; } ua, ub;
        const int e0 = (kk + half) >> 1;               // dword index
        ua.q[0] = *(const uint4*)(arow + e0);          // K +0..7
        ua.q[1] = *(const uint4*)(arow + e0 + 8);      // K +16..23
        ub.q[0] = *(const uint4*)(brow + e0);
        ub.q[1] = *(const uint4*)(brow + e0 + 8);
        c = __builtin_amdgcn_wmma_f32_16x16x32_bf16(
                /*neg_a=*/false, ua.v, /*neg_b=*/false, ub.v,
                /*c_mod=*/(short)0, c, /*reuse_a=*/false, /*reuse_b=*/false);
    }

    const int n = nTile * 16 + (lane & 15);
    if (n < U_) {
        const int mbase = mTile * 16 + ((lane < 16) ? 0 : 8);
        #pragma unroll
        for (int r = 0; r < 8; ++r)
            sim[(size_t)(mbase + r) * U_ + n] = c[r];
    }
}

// =====================================================================
// Kernel 3: per-batch vocab scan. sim[b] (64x600) staged to LDS as bf16
// (76.8 KB of the 320 KB/WGP). Each thread scans ~10 vocab words; packed
// (score-key<<32 | ~v) ds_max_u64 atomics give fused max+argmax per (l,e)
// with jnp's smallest-index tie-break. Then a 448-way packed reduction
// selects the best span and emits all 5 outputs.
// =====================================================================
__global__ __launch_bounds__(1024) void match_kernel(
    const float* __restrict__ sim,
    const int*   __restrict__ lengths,
    const int*   __restrict__ segs,     // [V][JMAX_]
    const int*   __restrict__ vlen,     // [V]
    float* __restrict__ out)
{
    __shared__ unsigned short      s_sim[L_ * U_];        // 76800 B
    __shared__ unsigned long long  s_best[L_ * EDIM];     // 3584 B
    __shared__ unsigned long long  s_bestAll;
    __shared__ int                 s_any;

    const int b   = blockIdx.x;
    const int tid = threadIdx.x;
    const float* simb = sim + (size_t)b * L_ * U_;

    for (int i = tid; i < L_ * U_; i += 1024)
        s_sim[i] = f2bf(simb[i]);

    const unsigned long long initpack =
        ((unsigned long long)encf(NEGV) << 32) | 0xFFFFFFFFull;   // decodes to v=0
    for (int i = tid; i < L_ * EDIM; i += 1024) s_best[i] = initpack;
    if (tid == 0) { s_bestAll = 0ull; s_any = 0; }
    __syncthreads();

    const int Lb = lengths[b];
    for (int v = tid; v < V_; v += 1024) {
        int len = vlen[v];
        if (len < MINLEN) len = MINLEN;
        if (len > JMAX_)  len = JMAX_;
        const int e = len - MINLEN;
        int seg[JMAX_];
        #pragma unroll
        for (int j = 0; j < JMAX_; ++j) seg[j] = segs[v * JMAX_ + j];
        const float rlen = 1.0f / (float)len;
        const int lmax = Lb - len;                        // inclusive last viable start
        const unsigned pay = 0xFFFFFFFFu ^ (unsigned)v;   // bigger == smaller v

        for (int l = 0; l <= lmax; ++l) {
            float sum = 0.0f;
            for (int j = 0; j < len; ++j)
                sum += bf2f(s_sim[(l + j) * U_ + seg[j]]);
            const float score = sum * rlen;
            const unsigned long long cand =
                ((unsigned long long)encf(score) << 32) | (unsigned long long)pay;
            if (cand > s_best[l * EDIM + e])              // test-and-test-and-set
                atomicMax(&s_best[l * EDIM + e], cand);
        }
    }
    __syncthreads();

    if (tid < L_ * EDIM) {
        const unsigned long long pk = s_best[tid];
        const unsigned key  = (unsigned)(pk >> 32);
        const float    val  = decf(key);
        const unsigned long long cand =
            ((unsigned long long)key << 32) |
            (unsigned long long)(0xFFFFFFFFu - (unsigned)tid);  // tie -> smallest flat
        atomicMax(&s_bestAll, cand);
        if (val > THRESH) atomicOr(&s_any, 1);
    }
    __syncthreads();

    if (tid == 0) {
        const unsigned long long pk = s_bestAll;
        const unsigned flat = 0xFFFFFFFFu - (unsigned)(pk & 0xFFFFFFFFull);
        const float    val  = decf((unsigned)(pk >> 32));
        const int l = (int)(flat / EDIM);
        const int e = (int)(flat % EDIM);
        const unsigned bestv =
            0xFFFFFFFFu ^ (unsigned)(s_best[flat] & 0xFFFFFFFFull);
        out[0 * B_ + b] = val;                              // best_scores
        out[1 * B_ + b] = (float)l;                         // best_starts
        out[2 * B_ + b] = (float)(l + e + MINLEN - 1);      // best_ends
        out[3 * B_ + b] = s_any ? 1.0f : 0.0f;              // any_matched
        out[4 * B_ + b] = (float)bestv;                     // best_vocab
    }
}

// =====================================================================
extern "C" void kernel_launch(void* const* d_in, const int* in_sizes, int n_in,
                              void* d_out, int out_size, void* d_ws, size_t ws_size,
                              hipStream_t stream)
{
    const float* x       = (const float*)d_in[0];   // [16,64,60]
    const float* uf      = (const float*)d_in[1];   // [600,60]
    const float* W       = (const float*)d_in[2];   // [60,256]
    const int*   lengths = (const int*)  d_in[3];   // [16]
    const int*   segs    = (const int*)  d_in[4];   // [10000,10]
    const int*   vlen    = (const int*)  d_in[5];   // [10000]
    float* out = (float*)d_out;

    char* ws = (char*)d_ws;
    const size_t wn_bytes = (size_t)NROWS * D_ * sizeof(unsigned short); // 512 KB
    const size_t un_bytes = (size_t)UPAD_ * D_ * sizeof(unsigned short); // 304 KB
    unsigned short* wn  = (unsigned short*)ws;
    unsigned short* un  = (unsigned short*)(ws + wn_bytes);
    float*          sim = (float*)(ws + wn_bytes + un_bytes);            // 2.4 MB

    embed_norm_kernel<<<NROWS + U_, 256, 0, stream>>>(x, uf, W, wn, un);
    sim_wmma_kernel<<<dim3(NROWS / 16, UPAD_ / 16), 32, 0, stream>>>(wn, un, sim);
    match_kernel<<<B_, 1024, 0, stream>>>(sim, lengths, segs, vlen, out);
}
